// VectorQuantizer_63952063038168
// MI455X (gfx1250) — matrix operand order repaired and checked by reading, not compiled, not run on hardware
//
#include <hip/hip_runtime.h>
#include <math.h>

#define NROWS   16384
#define EDIM    256
#define NCODES  1024
#define EPSF    1e-10f
#define LDSS    264   // padded bf16 row stride (16B pad) -> conflict-free ds_load_b128

typedef __attribute__((ext_vector_type(16))) __bf16 v16bf;
typedef __attribute__((ext_vector_type(8)))  float  v8f;

union V16 { uint4 q[2]; v16bf v; };

__device__ __forceinline__ unsigned short f32_to_bf16_bits(float x) {
  unsigned u = __builtin_bit_cast(unsigned, x);
  u += 0x7FFFu + ((u >> 16) & 1u);          // round-to-nearest-even
  return (unsigned short)(u >> 16);
}
__device__ __forceinline__ __bf16 f2bf(float x) {
  return __builtin_bit_cast(__bf16, f32_to_bf16_bits(x));
}
__device__ __forceinline__ float bf2f(__bf16 b) {
  unsigned u = ((unsigned)__builtin_bit_cast(unsigned short, b)) << 16;
  return __builtin_bit_cast(float, u);
}
__device__ __forceinline__ v8f wmma_bf16(v16bf a, v16bf b, v8f c) {
  return __builtin_amdgcn_wmma_f32_16x16x32_bf16(false, a, false, b, (short)0, c, false, false);
}

// ---------------------------------------------------------------- kPrep
// emb f32 -> bf16 hi/lo split, per-code squared norm, zero histogram.
__global__ __launch_bounds__(256) void kPrep(const float* __restrict__ emb,
                                             unsigned short* __restrict__ ehi,
                                             unsigned short* __restrict__ elo,
                                             float* __restrict__ e2,
                                             unsigned* __restrict__ counts) {
  int r = blockIdx.x * 256 + threadIdx.x;
  if (r >= NCODES) return;
  float s = 0.f;
  for (int k = 0; k < EDIM; ++k) {
    float x = emb[(size_t)r * EDIM + k];
    unsigned short hb = f32_to_bf16_bits(x);
    float hf = __builtin_bit_cast(float, ((unsigned)hb) << 16);
    ehi[(size_t)r * EDIM + k] = hb;
    elo[(size_t)r * EDIM + k] = f32_to_bf16_bits(x - hf);
    s += x * x;
  }
  e2[r] = s;
  counts[r] = 0u;
}

// ---------------------------------------------------------------- kDistArgmax
// Fused: logits = 2 z.e - |e|^2 + gumbel(u); per-row argmax over 1024 codes;
// then gather z_q = emb[idx], write z_q_st, partial sum of (z_q - z)^2, histogram.
__global__ __launch_bounds__(256) void kDistArgmax(const float* __restrict__ z,
                                                   const float* __restrict__ emb,
                                                   const float* __restrict__ u,
                                                   const unsigned short* __restrict__ ehi,
                                                   const unsigned short* __restrict__ elo,
                                                   const float* __restrict__ e2,
                                                   unsigned* __restrict__ counts,
                                                   float* __restrict__ diffPart,
                                                   float* __restrict__ outZq) {
  __shared__ unsigned short lds_hi[16 * LDSS];
  __shared__ unsigned short lds_lo[16 * LDSS];
  __shared__ int   lds_idx[128];
  __shared__ float lds_red[256];

  const int tid  = threadIdx.x;
  const int lane = tid & 31;
  const int wave = tid >> 5;
  const int h    = lane >> 4;     // K-half selector per WMMA layout
  const int rl   = lane & 15;     // row (A) / col (B) within tile
  const int blockBase = blockIdx.x * 128;
  const int rowBase   = blockBase + wave * 16;

  // ---- load this wave's 16 z rows as bf16 hi/lo A-operands (registers)
  v16bf Ahi[8], Alo[8];
  {
    const float* zrow = z + (size_t)(rowBase + rl) * EDIM;
    #pragma unroll
    for (int c = 0; c < 8; ++c) {
      const float* p = zrow + c * 32 + h * 8;
      float4 x0 = *(const float4*)(p + 0);
      float4 x1 = *(const float4*)(p + 4);
      float4 y0 = *(const float4*)(p + 16);
      float4 y1 = *(const float4*)(p + 20);
      float xs[16] = {x0.x,x0.y,x0.z,x0.w, x1.x,x1.y,x1.z,x1.w,
                      y0.x,y0.y,y0.z,y0.w, y1.x,y1.y,y1.z,y1.w};
      #pragma unroll
      for (int j = 0; j < 16; ++j) {
        __bf16 hb = f2bf(xs[j]);
        Ahi[c][j] = hb;
        Alo[c][j] = f2bf(xs[j] - bf2f(hb));
      }
    }
  }

  float bs[8];
  int   bi[8];
  #pragma unroll
  for (int v = 0; v < 8; ++v) { bs[v] = -3.402823466e38f; bi[v] = 0; }

  const int srr = tid >> 4;          // staging: row in tile
  const int skb = (tid & 15) << 4;   // staging: element offset (16 bf16 per thread)

  for (int t = 0; t < 64; ++t) {
    __syncthreads();
    { // stage emb tile t (16 codes x 256, hi+lo) into LDS, coalesced
      size_t src = (size_t)(t * 16 + srr) * EDIM + skb;
      const uint4* sh = (const uint4*)(ehi + src);
      const uint4* sl = (const uint4*)(elo + src);
      uint4 h0 = sh[0], h1 = sh[1], l0 = sl[0], l1 = sl[1];
      uint4* dh = (uint4*)(&lds_hi[srr * LDSS + skb]);
      uint4* dl = (uint4*)(&lds_lo[srr * LDSS + skb]);
      dh[0] = h0; dh[1] = h1; dl[0] = l0; dl[1] = l1;
      if (t + 1 < 64) { // pull next tile toward L2/L0 (global_prefetch_b8)
        __builtin_prefetch(ehi + src + (size_t)16 * EDIM, 0, 1);
        __builtin_prefetch(elo + src + (size_t)16 * EDIM, 0, 1);
      }
    }
    __syncthreads();

    v8f C = {0.f,0.f,0.f,0.f,0.f,0.f,0.f,0.f};
    #pragma unroll
    for (int c = 0; c < 8; ++c) {
      int off = c * 32 + h * 8;      // B layout: lane=col, K {off..off+7} U {off+16..off+23}
      V16 bh, bl;
      const uint4* ph = (const uint4*)(&lds_hi[rl * LDSS + off]);
      const uint4* pl = (const uint4*)(&lds_lo[rl * LDSS + off]);
      bh.q[0] = ph[0]; bh.q[1] = ph[2];
      bl.q[0] = pl[0]; bl.q[1] = pl[2];
      C = wmma_bf16(Ahi[c], bh.v, C);   // hi*hi
      C = wmma_bf16(Ahi[c], bl.v, C);   // hi*lo
      C = wmma_bf16(Alo[c], bh.v, C);   // lo*hi
    }

    // epilogue: gumbel + running argmax. C[v] = z_row(v+8h) . e_col(n)
    int n = t * 16 + rl;
    float e2n = e2[n];
    #pragma unroll
    for (int v = 0; v < 8; ++v) {
      int r = v + 8 * h;
      float uv = u[(size_t)(rowBase + r) * NCODES + n];
      float g  = -logf(-logf(uv + EPSF) + EPSF);
      float score = 2.0f * C[v] - e2n + g;
      if (score > bs[v]) { bs[v] = score; bi[v] = n; }   // strict > == first max
    }
  }

  // ---- cross-lane argmax within each 16-lane half (ties -> smaller idx)
  #pragma unroll
  for (int v = 0; v < 8; ++v) {
    float s = bs[v]; int i = bi[v];
    #pragma unroll
    for (int m = 1; m < 16; m <<= 1) {
      float s2 = __shfl_xor(s, m, 16);
      int   i2 = __shfl_xor(i, m, 16);
      if (s2 > s || (s2 == s && i2 < i)) { s = s2; i = i2; }
    }
    if (rl == 0) lds_idx[wave * 16 + v + 8 * h] = i;
  }
  __syncthreads();

  // ---- gather z_q = emb[idx], write output, accumulate (z_q - z)^2
  float sacc = 0.f;
  for (int r = 0; r < 128; ++r) {
    int rowg = blockBase + r;
    int code = lds_idx[r];
    float ev = emb[(size_t)code * EDIM + tid];
    float zv = z[(size_t)rowg * EDIM + tid];
    float dd = ev - zv;
    sacc += dd * dd;
    outZq[(size_t)rowg * EDIM + tid] = ev;
  }
  if (tid < 128) atomicAdd(&counts[lds_idx[tid]], 1u);

  lds_red[tid] = sacc; __syncthreads();
  for (int s = 128; s > 0; s >>= 1) {
    if (tid < s) lds_red[tid] += lds_red[tid + s];
    __syncthreads();
  }
  if (tid == 0) diffPart[blockIdx.x] = lds_red[0];
}

// ---------------------------------------------------------------- kOrtho
// Sum over (emb@emb^T - I)^2, 16x16 WMMA tiles, 4 tiles per wave.
__global__ __launch_bounds__(256) void kOrtho(const unsigned short* __restrict__ ehi,
                                              const unsigned short* __restrict__ elo,
                                              float* __restrict__ orthoPart) {
  const int tid  = threadIdx.x;
  const int lane = tid & 31;
  const int wave = tid >> 5;
  const int h    = lane >> 4;
  const int rl   = lane & 15;
  const int W    = blockIdx.x * 8 + wave;   // 1024 waves, 4096 tiles

  float acc = 0.f;
  for (int tt = 0; tt < 4; ++tt) {
    int tile = W * 4 + tt;
    int mi = tile >> 6, ni = tile & 63;
    size_t arow = (size_t)(mi * 16 + rl) * EDIM;
    size_t brow = (size_t)(ni * 16 + rl) * EDIM;
    v8f C = {0.f,0.f,0.f,0.f,0.f,0.f,0.f,0.f};
    #pragma unroll
    for (int c = 0; c < 8; ++c) {
      int ka = c * 32 + h * 8;   // A layout: {ka..ka+7} U {ka+16..ka+23}
      int kb = c * 32 + h * 8;   // B layout: {kb..kb+7} U {kb+16..kb+23}
      V16 ah, al, bh, bl;
      ah.q[0] = *(const uint4*)(ehi + arow + ka);
      ah.q[1] = *(const uint4*)(ehi + arow + ka + 16);
      al.q[0] = *(const uint4*)(elo + arow + ka);
      al.q[1] = *(const uint4*)(elo + arow + ka + 16);
      bh.q[0] = *(const uint4*)(ehi + brow + kb);
      bh.q[1] = *(const uint4*)(ehi + brow + kb + 16);
      bl.q[0] = *(const uint4*)(elo + brow + kb);
      bl.q[1] = *(const uint4*)(elo + brow + kb + 16);
      C = wmma_bf16(ah.v, bh.v, C);
      C = wmma_bf16(ah.v, bl.v, C);
      C = wmma_bf16(al.v, bh.v, C);
    }
    #pragma unroll
    for (int v = 0; v < 8; ++v) {
      int m = mi * 16 + v + 8 * h;
      int n = ni * 16 + rl;
      float gv = C[v] - (m == n ? 1.0f : 0.0f);
      acc += gv * gv;
    }
  }
  #pragma unroll
  for (int m = 1; m < 32; m <<= 1) acc += __shfl_xor(acc, m, 32);
  if (lane == 0) orthoPart[W] = acc;
}

// ---------------------------------------------------------------- kFinal
__global__ __launch_bounds__(256) void kFinal(const unsigned* __restrict__ counts,
                                              const float* __restrict__ diffPart,
                                              const float* __restrict__ orthoPart,
                                              float* __restrict__ out) {
  __shared__ float red[256];
  const int tid = threadIdx.x;

  float ent = 0.f;
  for (int b = tid; b < NCODES; b += 256) {
    float p = (float)counts[b] * (1.0f / (float)NROWS);
    ent += p * logf(p + EPSF);
  }
  red[tid] = ent; __syncthreads();
  for (int s = 128; s > 0; s >>= 1) { if (tid < s) red[tid] += red[tid + s]; __syncthreads(); }
  float entropy = red[0]; __syncthreads();

  float dl = (tid < 128) ? diffPart[tid] : 0.f;
  red[tid] = dl; __syncthreads();
  for (int s = 128; s > 0; s >>= 1) { if (tid < s) red[tid] += red[tid + s]; __syncthreads(); }
  float S = red[0]; __syncthreads();

  float ol = 0.f;
  for (int b = tid; b < 1024; b += 256) ol += orthoPart[b];
  red[tid] = ol; __syncthreads();
  for (int s = 128; s > 0; s >>= 1) { if (tid < s) red[tid] += red[tid + s]; __syncthreads(); }
  float O = red[0];

  if (tid == 0) {
    float m = S / ((float)NROWS * (float)EDIM);           // mean (z_q - z)^2
    float ortho = (O / (1024.f * 1024.f)) / (1024.f * 1024.f);
    float loss = m * (1.0f + 0.25f * 0.25f) + 0.1f * ortho;
    out[0] = loss;
    out[1 + (size_t)NROWS * EDIM] = expf(-entropy);
  }
}

// ---------------------------------------------------------------- launcher
extern "C" void kernel_launch(void* const* d_in, const int* in_sizes, int n_in,
                              void* d_out, int out_size, void* d_ws, size_t ws_size,
                              hipStream_t stream) {
  (void)in_sizes; (void)n_in; (void)out_size; (void)ws_size;
  const float* z   = (const float*)d_in[0];   // [16,1024,256]
  const float* emb = (const float*)d_in[1];   // [1024,256]
  const float* u   = (const float*)d_in[2];   // [16384,1024]
  float* out = (float*)d_out;                 // [loss | z_q_st | perplexity]

  char* ws = (char*)d_ws;
  unsigned short* ehi    = (unsigned short*)(ws);              // 512 KB
  unsigned short* elo    = (unsigned short*)(ws + 524288);     // 512 KB
  float*          e2     = (float*)(ws + 1048576);             // 4 KB
  unsigned*       counts = (unsigned*)(ws + 1052672);          // 4 KB
  float*          diffP  = (float*)(ws + 1056768);             // 512 B
  float*          orthoP = (float*)(ws + 1057280);             // 4 KB

  kPrep<<<dim3(4), dim3(256), 0, stream>>>(emb, ehi, elo, e2, counts);
  kDistArgmax<<<dim3(128), dim3(256), 0, stream>>>(z, emb, u, ehi, elo, e2,
                                                   counts, diffP, out + 1);
  kOrtho<<<dim3(128), dim3(256), 0, stream>>>(ehi, elo, orthoP);
  kFinal<<<dim3(1), dim3(256), 0, stream>>>(counts, diffP, orthoP, out);
}
